// DSTFPE_24704651887215
// MI455X (gfx1250) — compile-verified, matrix operand order checked by reading, not run on hardware
//
#include <hip/hip_runtime.h>
#include <hip/hip_bf16.h>

// ---------------- model constants ----------------
#define E_   512
#define H_   8
#define D_   64
#define RANK_ 256
#define EXP_  1536
#define HEXP_ 768
#define B_    8
#define S_    2048
#define M_    16384      // B_*S_
#define OUT_  15

#define ACT_NONE  0
#define ACT_RELU6 1
#define ACT_RELU  2

// ---------------- WMMA types ----------------
typedef __bf16 bf16x16 __attribute__((ext_vector_type(16)));
typedef float  v8f     __attribute__((ext_vector_type(8)));

__device__ __forceinline__ unsigned short f2bf(float f) {
    unsigned u = __float_as_uint(f);
    u += 0x7FFFu + ((u >> 16) & 1u);        // round-to-nearest-even
    return (unsigned short)(u >> 16);
}

__device__ __forceinline__ unsigned pack2bf(float a, float b) {
    return (unsigned)f2bf(a) | ((unsigned)f2bf(b) << 16);
}

__device__ __forceinline__ bf16x16 load_frag(const unsigned short* p) {
    uint4 t[2];
    t[0] = *reinterpret_cast<const uint4*>(p);        // K 0..7  of lane's group
    t[1] = *reinterpret_cast<const uint4*>(p + 16);   // K 16..23 of lane's group
    bf16x16 r;
    __builtin_memcpy(&r, t, 32);
    return r;
}

// ---------------- generic fused WMMA GEMM ----------------
// C[M,N] = act(A[M,K] @ W[K,N] + bias), f32 in/out, bf16 tensor-core inputs.
// Double-buffered LDS staging; vectorized (b128) tile loads.
// Batched via blockIdx.z with split (outer,inner) strides.
#define BM 128
#define BN 128
#define BK 32
#define LDT 40   // padded bf16 k-stride: 80B rows keep ds_load_b128 alignment

// stage one 128x32 A tile and one 32x128 B tile (transposed) into LDS as bf16
__device__ __forceinline__
void stage_tiles(const float* __restrict__ A, int ldA,
                 const float* __restrict__ Wt, int ldW,
                 int N, int row0, int col0, int k0, int tid,
                 unsigned short* __restrict__ as_, unsigned short* __restrict__ bs_) {
    // ---- A: 1024 float4 loads, packed bf16 ds_store_b64 ----
#pragma unroll
    for (int i = 0; i < 4; ++i) {
        int e = tid + i * 256;
        int r = e >> 3, c4 = (e & 7) * 4;             // ldA,K multiples of 4/32 -> aligned, in range
        float4 f = *reinterpret_cast<const float4*>(&A[(size_t)(row0 + r) * ldA + k0 + c4]);
        uint2 u;
        u.x = pack2bf(f.x, f.y);
        u.y = pack2bf(f.z, f.w);
        *reinterpret_cast<uint2*>(&as_[r * LDT + c4]) = u;
    }
    // ---- B: transposed store Bs[n][k]; vector path when N%4==0 ----
    if ((N & 3) == 0) {
#pragma unroll
        for (int i = 0; i < 4; ++i) {
            int e = tid + i * 256;
            int kk = e >> 5, n4 = (e & 31) * 4;
            int gc = col0 + n4;
            float4 f = make_float4(0.f, 0.f, 0.f, 0.f);
            if (gc < N)
                f = *reinterpret_cast<const float4*>(&Wt[(size_t)(k0 + kk) * ldW + gc]);
            bs_[(n4 + 0) * LDT + kk] = f2bf(f.x);
            bs_[(n4 + 1) * LDT + kk] = f2bf(f.y);
            bs_[(n4 + 2) * LDT + kk] = f2bf(f.z);
            bs_[(n4 + 3) * LDT + kk] = f2bf(f.w);
        }
    } else {
#pragma unroll
        for (int i = 0; i < 16; ++i) {
            int e = tid + i * 256;
            int n = e & 127, kk = e >> 7;
            int gc = col0 + n;
            float f = (gc < N) ? Wt[(size_t)(k0 + kk) * ldW + gc] : 0.0f;
            bs_[n * LDT + kk] = f2bf(f);
        }
    }
}

__global__ __launch_bounds__(256)
void gemm_wmma_kernel(const float* __restrict__ A, int ldA,
                      const float* __restrict__ Wt, int ldW,
                      const float* __restrict__ bias,
                      float* __restrict__ C, int ldC,
                      int M, int N, int K, int act,
                      int bInner,
                      long long sAo, long long sAi,
                      long long sWo, long long sWi,
                      long long sCo, long long sCi)
{
    const int z  = blockIdx.z;
    const int zo = z / bInner, zi = z % bInner;
    A  += zo * sAo + zi * sAi;
    Wt += zo * sWo + zi * sWi;
    C  += zo * sCo + zi * sCi;

    __shared__ __align__(16) unsigned short As[2][BM * LDT];
    __shared__ __align__(16) unsigned short Bs[2][BN * LDT];

    const int tid  = threadIdx.x;
    const int lane = tid & 31;
    const int wave = tid >> 5;
    const int row0 = blockIdx.y * BM;
    const int col0 = blockIdx.x * BN;
    const int wm   = (wave & 3) * 32;   // 4 waves along M (32 rows each)
    const int wn   = (wave >> 2) * 64;  // 2 waves along N (64 cols each)

    v8f acc[2][4];
#pragma unroll
    for (int i = 0; i < 2; ++i)
#pragma unroll
        for (int j = 0; j < 4; ++j) acc[i][j] = (v8f){};

    const int kb = (lane >> 4) * 8;     // K-group base per lane half
    const int lr = lane & 15;
    const int nk = K / BK;

    stage_tiles(A, ldA, Wt, ldW, N, row0, col0, 0, tid, As[0], Bs[0]);
    __syncthreads();

    for (int kt = 0; kt < nk; ++kt) {
        const int cur = kt & 1;
        if (kt + 1 < nk)
            stage_tiles(A, ldA, Wt, ldW, N, row0, col0, (kt + 1) * BK, tid,
                        As[cur ^ 1], Bs[cur ^ 1]);

        const unsigned short* as_ = As[cur];
        const unsigned short* bs_ = Bs[cur];
        bf16x16 af[2], bfrag[4];
#pragma unroll
        for (int ti = 0; ti < 2; ++ti)
            af[ti] = load_frag(&as_[(wm + ti * 16 + lr) * LDT + kb]);
#pragma unroll
        for (int tj = 0; tj < 4; ++tj)
            bfrag[tj] = load_frag(&bs_[(wn + tj * 16 + lr) * LDT + kb]);

#pragma unroll
        for (int ti = 0; ti < 2; ++ti)
#pragma unroll
            for (int tj = 0; tj < 4; ++tj)
                acc[ti][tj] = __builtin_amdgcn_wmma_f32_16x16x32_bf16(
                    false, af[ti], false, bfrag[tj],
                    (short)0, acc[ti][tj], false, false);
        __syncthreads();
    }

    // ---- epilogue: bias + activation, guarded on N ----
#pragma unroll
    for (int tj = 0; tj < 4; ++tj) {
        int colx = col0 + wn + tj * 16 + (lane & 15);
        if (colx >= N) continue;
        float bv = bias ? bias[colx] : 0.0f;
#pragma unroll
        for (int ti = 0; ti < 2; ++ti) {
            int rbase = row0 + wm + ti * 16 + 8 * (lane >> 4);
#pragma unroll
            for (int r = 0; r < 8; ++r) {
                float v = acc[ti][tj][r] + bv;
                if (act == ACT_RELU6) v = fminf(fmaxf(v, 0.0f), 6.0f);
                else if (act == ACT_RELU) v = fmaxf(v, 0.0f);
                C[(size_t)(rbase + r) * ldC + colx] = v;
            }
        }
    }
}

// ---------------- elementwise / fused kernels ----------------

// x = elu(x)+1  (x>0 ? x+1 : exp(x)), in place
__global__ void elu1_kernel(float* x, int n) {
    int i = blockIdx.x * blockDim.x + threadIdx.x;
    if (i < n) { float v = x[i]; x[i] = (v > 0.0f) ? v + 1.0f : __expf(v); }
}

// ksum[b,h,d] = sum_s k[b,s,h,d]    (k laid out [B,S,E])
__global__ void ksum_kernel(const float* __restrict__ k, float* __restrict__ ks) {
    int i = blockIdx.x * blockDim.x + threadIdx.x;     // 4096 = B*H*D
    if (i >= B_ * E_) return;
    int b = i >> 9, hd = i & (E_ - 1);
    float s = 0.0f;
    for (int ss = 0; ss < S_; ++ss)
        s += k[((size_t)(b * S_ + ss) * E_) + hd];
    ks[i] = s;
}

// kv[b,h,d,e] = sum_s k[b,s,h,d]*v[b,s,h,e], one block per (b,h)
__global__ __launch_bounds__(256)
void kv_kernel(const float* __restrict__ k, const float* __restrict__ v,
               float* __restrict__ kv) {
    int bh = blockIdx.x, b = bh >> 3, h = bh & 7;
    __shared__ float ks[32][D_];
    __shared__ float vs[32][D_];
    int tid = threadIdx.x;
    int e = tid & 63, d4 = tid >> 6;         // d4 in 0..3 -> 16 d's each
    float acc[16];
#pragma unroll
    for (int i = 0; i < 16; ++i) acc[i] = 0.0f;
    for (int s0 = 0; s0 < S_; s0 += 32) {
#pragma unroll
        for (int i = 0; i < 8; ++i) {
            int t = tid + i * 256, ss = t >> 6, dd = t & 63;
            size_t gi = ((size_t)(b * S_ + s0 + ss) * E_) + h * D_ + dd;
            ks[ss][dd] = k[gi];
            vs[ss][dd] = v[gi];
        }
        __syncthreads();
        for (int ss = 0; ss < 32; ++ss) {
            float vv = vs[ss][e];
#pragma unroll
            for (int i = 0; i < 16; ++i) acc[i] += ks[ss][d4 * 16 + i] * vv;
        }
        __syncthreads();
    }
#pragma unroll
    for (int i = 0; i < 16; ++i)
        kv[(size_t)bh * (D_ * D_) + (d4 * 16 + i) * D_ + e] = acc[i];
}

// den[row,h] = sum_d q[row, h*64+d] * ksum[b,h,d]
__global__ void den_kernel(const float* __restrict__ q, const float* __restrict__ ks,
                           float* __restrict__ den) {
    int i = blockIdx.x * blockDim.x + threadIdx.x;     // M_*H_
    if (i >= M_ * H_) return;
    int h = i & 7, row = i >> 3, b = row >> 11;
    const float* qp = q + (size_t)row * E_ + h * D_;
    const float* kp = ks + (b * H_ + h) * D_;
    float s = 0.0f;
    for (int d = 0; d < D_; ++d) s += qp[d] * kp[d];
    den[i] = s;
}

// num /= (den + 1e-6), in place on [M, E]
__global__ void divden_kernel(float* num, const float* __restrict__ den, int n) {
    int i = blockIdx.x * blockDim.x + threadIdx.x;
    if (i >= n) return;
    int row = i >> 9, h = (i & (E_ - 1)) >> 6;
    num[i] = num[i] / (den[row * H_ + h] + 1e-6f);
}

// softmax over last dim of [rows, 2], in place
__global__ void softmax2_kernel(float* x, int rows) {
    int r = blockIdx.x * blockDim.x + threadIdx.x;
    if (r >= rows) return;
    float a = x[r * 2], b = x[r * 2 + 1];
    float m = fmaxf(a, b);
    float p = __expf(a - m), q = __expf(b - m);
    float s = p + q;
    x[r * 2] = p / s; x[r * 2 + 1] = q / s;
}

// y = act( LN(x + gate*res) * g + b ), in place on x. res/gate optional.
__global__ __launch_bounds__(256)
void ln_kernel(float* __restrict__ x, const float* __restrict__ res, int ldRes,
               const float* __restrict__ gate, int gcol,
               const float* __restrict__ gw, const float* __restrict__ bw, int act) {
    __shared__ float s1[256];
    __shared__ float s2[256];
    int row = blockIdx.x, tid = threadIdx.x;
    float sc = gate ? gate[row * 2 + gcol] : 1.0f;
    int c0 = tid, c1 = tid + 256;
    float v0 = x[(size_t)row * E_ + c0];
    float v1 = x[(size_t)row * E_ + c1];
    if (res) {
        v0 += sc * res[(size_t)row * ldRes + c0];
        v1 += sc * res[(size_t)row * ldRes + c1];
    }
    s1[tid] = v0 + v1;
    s2[tid] = v0 * v0 + v1 * v1;
    __syncthreads();
    for (int o = 128; o > 0; o >>= 1) {
        if (tid < o) { s1[tid] += s1[tid + o]; s2[tid] += s2[tid + o]; }
        __syncthreads();
    }
    float mean = s1[0] * (1.0f / E_);
    float var  = s2[0] * (1.0f / E_) - mean * mean;
    float rstd = rsqrtf(var + 1e-5f);
    float y0 = (v0 - mean) * rstd * gw[c0] + bw[c0];
    float y1 = (v1 - mean) * rstd * gw[c1] + bw[c1];
    if (act == ACT_RELU) { y0 = fmaxf(y0, 0.0f); y1 = fmaxf(y1, 0.0f); }
    x[(size_t)row * E_ + c0] = y0;
    x[(size_t)row * E_ + c1] = y1;
}

// depthwise conv3 (pad 1, over s) + bias + BN(eval) + relu6;  [M, EXP] layout
__global__ void dwconv_bn_kernel(const float* __restrict__ in, float* __restrict__ out,
                                 const float* __restrict__ dw, const float* __restrict__ dwb,
                                 const float* __restrict__ bng, const float* __restrict__ bnb,
                                 const float* __restrict__ bnm, const float* __restrict__ bnv) {
    int i = blockIdx.x * blockDim.x + threadIdx.x;
    if (i >= M_ * EXP_) return;
    int c = i % EXP_;
    int rs = i / EXP_;
    int s = rs & (S_ - 1);
    float xm = (s > 0)      ? in[(size_t)(rs - 1) * EXP_ + c] : 0.0f;
    float x0 = in[(size_t)rs * EXP_ + c];
    float xp = (s < S_ - 1) ? in[(size_t)(rs + 1) * EXP_ + c] : 0.0f;
    float a = xm * dw[c * 3] + x0 * dw[c * 3 + 1] + xp * dw[c * 3 + 2] + dwb[c];
    a = (a - bnm[c]) * rsqrtf(bnv[c] + 1e-5f) * bng[c] + bnb[c];
    out[i] = fminf(fmaxf(a, 0.0f), 6.0f);
}

// out[row*1024 + ofs + c] = in[row*512 + c]  (concat staging)
__global__ void copy_col_kernel(const float* __restrict__ in, float* __restrict__ out, int ofs) {
    int i = blockIdx.x * blockDim.x + threadIdx.x;
    if (i >= M_ * E_) return;
    int row = i >> 9, c = i & (E_ - 1);
    out[(size_t)row * (2 * E_) + ofs + c] = in[i];
}

// ---------------- host orchestration ----------------

static inline void gemm(hipStream_t st, const float* A, int ldA,
                        const float* Wt, int ldW, const float* bias,
                        float* C, int ldC, int M, int N, int K, int act,
                        int batch = 1, int bInner = 1,
                        long long sAo = 0, long long sAi = 0,
                        long long sWo = 0, long long sWi = 0,
                        long long sCo = 0, long long sCi = 0) {
    dim3 grid((N + BN - 1) / BN, M / BM, batch);
    gemm_wmma_kernel<<<grid, 256, 0, st>>>(A, ldA, Wt, ldW, bias, C, ldC,
                                           M, N, K, act, bInner,
                                           sAo, sAi, sWo, sWi, sCo, sCi);
}

static inline void launch_ln(hipStream_t st, float* x, const float* res, int ldRes,
                             const float* gate, int gcol,
                             const float* gw, const float* bw, int act) {
    ln_kernel<<<M_, 256, 0, st>>>(x, res, ldRes, gate, gcol, gw, bw, act);
}

struct Scratch {
    float *Xb, *Xl, *G, *Qb, *Kb, *Vb, *NUM, *T1, *T2, *DEN, *KS, *KV, *GT3;
};

#define IN(i) ((const float*)d_in[(i)])

// linear attention: q from xq, k/v from xkv; result -> dst (row stride ldDst)
static void run_attn(hipStream_t st, void* const* d_in, int pb,
                     const float* xq, const float* xkv,
                     float* dst, int ldDst, const Scratch& w) {
    const int NE = M_ * E_;
    // low-rank q/k/v projections (T1 front used as down-proj temp)
    gemm(st, xq,  E_, IN(pb + 0), RANK_, nullptr,      w.T1, RANK_, M_, RANK_, E_, ACT_NONE);
    gemm(st, w.T1, RANK_, IN(pb + 1), E_, IN(pb + 2),  w.Qb, E_,    M_, E_, RANK_, ACT_NONE);
    gemm(st, xkv, E_, IN(pb + 3), RANK_, nullptr,      w.T1, RANK_, M_, RANK_, E_, ACT_NONE);
    gemm(st, w.T1, RANK_, IN(pb + 4), E_, IN(pb + 5),  w.Kb, E_,    M_, E_, RANK_, ACT_NONE);
    gemm(st, xkv, E_, IN(pb + 6), RANK_, nullptr,      w.T1, RANK_, M_, RANK_, E_, ACT_NONE);
    gemm(st, w.T1, RANK_, IN(pb + 7), E_, IN(pb + 8),  w.Vb, E_,    M_, E_, RANK_, ACT_NONE);
    // elu(x)+1 feature map on q,k
    elu1_kernel<<<NE / 256, 256, 0, st>>>(w.Qb, NE);
    elu1_kernel<<<NE / 256, 256, 0, st>>>(w.Kb, NE);
    // ksum, kv, den
    ksum_kernel<<<(B_ * E_) / 256, 256, 0, st>>>(w.Kb, w.KS);
    kv_kernel<<<B_ * H_, 256, 0, st>>>(w.Kb, w.Vb, w.KV);
    den_kernel<<<(M_ * H_) / 256, 256, 0, st>>>(w.Qb, w.KS, w.DEN);
    // num = q @ kv  : batched WMMA GEMM, batch = B*H (outer b, inner h)
    gemm(st, w.Qb, E_, w.KV, D_, nullptr, w.NUM, E_,
         S_, D_, D_, ACT_NONE,
         B_ * H_, H_,
         (long long)S_ * E_, D_,                       // A: b-> +S*E, h-> +64
         (long long)H_ * D_ * D_, (long long)D_ * D_,  // W: kv[b,h]
         (long long)S_ * E_, D_);                      // C: same layout as q
    divden_kernel<<<NE / 256, 256, 0, st>>>(w.NUM, w.DEN, NE);
    // output projection
    gemm(st, w.NUM, E_, IN(pb + 9), E_, IN(pb + 10), dst, ldDst, M_, E_, E_, ACT_NONE);
}

// FFN with fused dwconv+BN+relu6; LN(x + ffn(x)) in place on x
static void run_ffn(hipStream_t st, void* const* d_in, int fb,
                    float* x, const float* ng, const float* nb, const Scratch& w) {
    gemm(st, x,    E_,    IN(fb + 0),  HEXP_, IN(fb + 1),  w.T1, HEXP_, M_, HEXP_, E_,    ACT_RELU6);
    gemm(st, w.T1, HEXP_, IN(fb + 2),  EXP_,  IN(fb + 3),  w.T2, EXP_,  M_, EXP_,  HEXP_, ACT_RELU6);
    dwconv_bn_kernel<<<(M_ * EXP_) / 256, 256, 0, st>>>(
        w.T2, w.T1, IN(fb + 4), IN(fb + 5), IN(fb + 6), IN(fb + 7), IN(fb + 8), IN(fb + 9));
    gemm(st, w.T1, EXP_,  IN(fb + 10), HEXP_, IN(fb + 11), w.T2, HEXP_, M_, HEXP_, EXP_,  ACT_RELU6);
    gemm(st, w.T2, HEXP_, IN(fb + 12), E_,    IN(fb + 13), w.Qb, E_,    M_, E_,    HEXP_, ACT_NONE);
    launch_ln(st, x, w.Qb, E_, nullptr, 0, ng, nb, ACT_NONE);
}

extern "C" void kernel_launch(void* const* d_in, const int* in_sizes, int n_in,
                              void* d_out, int out_size, void* d_ws, size_t ws_size,
                              hipStream_t stream) {
    (void)in_sizes; (void)n_in; (void)out_size; (void)ws_size;
    const size_t NE = (size_t)M_ * E_;

    float* ws = (float*)d_ws;
    Scratch w;
    w.Xb  = ws;               ws += NE;
    w.Xl  = ws;               ws += NE;
    w.G   = ws;               ws += 2 * NE;          // concat buffer [M, 1024]
    w.Qb  = ws;               ws += NE;
    w.Kb  = ws;               ws += NE;
    w.Vb  = ws;               ws += NE;
    w.NUM = ws;               ws += NE;
    w.T1  = ws;               ws += (size_t)M_ * EXP_;
    w.T2  = ws;               ws += (size_t)M_ * EXP_;
    w.DEN = ws;               ws += (size_t)M_ * H_;
    w.KS  = ws;               ws += (size_t)B_ * E_;
    w.KV  = ws;               ws += (size_t)B_ * H_ * D_ * D_;
    w.GT3 = ws;               ws += (size_t)M_ * 2;

    hipMemcpyAsync(w.Xb, d_in[0], NE * sizeof(float), hipMemcpyDeviceToDevice, stream);
    hipMemcpyAsync(w.Xl, d_in[1], NE * sizeof(float), hipMemcpyDeviceToDevice, stream);

    for (int layer = 0; layer < 3; ++layer) {
        const int base = 2 + layer * 90;
        const int saB = base + 0, saL = base + 11, caBL = base + 22, caLB = base + 33;
        const int ffB = base + 44, ffL = base + 58, gg = base + 72, nn = base + 78;

        // self attention + LN1 (G used as attention-output scratch, ld 512)
        run_attn(stream, d_in, saB, w.Xb, w.Xb, w.G, E_, w);
        launch_ln(stream, w.Xb, w.G, E_, nullptr, 0, IN(nn + 0), IN(nn + 1), ACT_NONE);
        run_attn(stream, d_in, saL, w.Xl, w.Xl, w.G, E_, w);
        launch_ln(stream, w.Xl, w.G, E_, nullptr, 0, IN(nn + 2), IN(nn + 3), ACT_NONE);

        // cross attention -> concat halves of G (ld 1024)
        run_attn(stream, d_in, caBL, w.Xb, w.Xl, w.G,      2 * E_, w);  // cb
        run_attn(stream, d_in, caLB, w.Xl, w.Xb, w.G + E_, 2 * E_, w);  // cl

        // gate MLP + softmax
        gemm(stream, w.G,  2 * E_, IN(gg + 0), E_ / 2, IN(gg + 1), w.T1, E_ / 2, M_, E_ / 2, 2 * E_, ACT_RELU6);
        gemm(stream, w.T1, E_ / 2, IN(gg + 2), E_ / 4, IN(gg + 3), w.T2, E_ / 4, M_, E_ / 4, E_ / 2, ACT_RELU6);
        gemm(stream, w.T2, E_ / 4, IN(gg + 4), 2,      IN(gg + 5), w.GT3, 2,     M_, 2,      E_ / 4, ACT_NONE);
        softmax2_kernel<<<M_ / 256, 256, 0, stream>>>(w.GT3, M_);

        // gated residual + LN2
        launch_ln(stream, w.Xb, w.G,      2 * E_, w.GT3, 0, IN(nn + 4), IN(nn + 5), ACT_NONE);
        launch_ln(stream, w.Xl, w.G + E_, 2 * E_, w.GT3, 1, IN(nn + 6), IN(nn + 7), ACT_NONE);

        // FFN + LN3
        run_ffn(stream, d_in, ffB, w.Xb, IN(nn + 8),  IN(nn + 9),  w);
        run_ffn(stream, d_in, ffL, w.Xl, IN(nn + 10), IN(nn + 11), w);
    }

    // fusion head
    const int fu = 2 + 3 * 90;      // 272
    const int rg = fu + 8;          // 280
    copy_col_kernel<<<(M_ * E_) / 256, 256, 0, stream>>>(w.Xb, w.G, 0);
    copy_col_kernel<<<(M_ * E_) / 256, 256, 0, stream>>>(w.Xl, w.G, E_);
    gemm(stream, w.G,  2 * E_, IN(fu + 0), E_,     IN(fu + 1), w.Qb, E_,     M_, E_,     2 * E_, ACT_RELU6);
    gemm(stream, w.Qb, E_,     IN(fu + 2), E_ / 2, IN(fu + 3), w.T1, E_ / 2, M_, E_ / 2, E_,     ACT_RELU6);
    gemm(stream, w.T1, E_ / 2, IN(fu + 4), E_,     IN(fu + 5), w.Kb, E_,     M_, E_,     E_ / 2, ACT_NONE);
    launch_ln(stream, w.Kb, nullptr, 0, nullptr, 0, IN(fu + 6), IN(fu + 7), ACT_RELU);

    // regression head -> d_out [M, 15]
    float* out = (float*)d_out;
    gemm(stream, w.Kb, E_,     IN(rg + 0), E_ / 2, IN(rg + 1), w.T1, E_ / 2, M_, E_ / 2, E_,     ACT_RELU6);
    gemm(stream, w.T1, E_ / 2, IN(rg + 2), E_ / 4, IN(rg + 3), w.T2, E_ / 4, M_, E_ / 4, E_ / 2, ACT_RELU6);
    gemm(stream, w.T2, E_ / 4, IN(rg + 4), E_ / 8, IN(rg + 5), w.T1, E_ / 8, M_, E_ / 8, E_ / 4, ACT_RELU6);
    gemm(stream, w.T1, E_ / 8, IN(rg + 6), OUT_,   IN(rg + 7), out,  OUT_,   M_, OUT_,   E_ / 8, ACT_NONE);
}
